// HashLinear_30528627540665
// MI455X (gfx1250) — matrix operand order repaired and checked by reading, not compiled, not run on hardware
//
#include <hip/hip_runtime.h>

typedef __attribute__((ext_vector_type(16))) _Float16 v16h;
typedef __attribute__((ext_vector_type(8)))  _Float16 v8h;
typedef __attribute__((ext_vector_type(4)))  _Float16 v4h;
typedef __attribute__((ext_vector_type(8)))  float    v8f;

#define M_DIM 8192
#define N_DIM 4096
#define K_DIM 4096

#define BM 128
#define BN 128
#define KSTEP 32
#define LDA 40   // halfs per LDS row: 32 + 8 pad (80B = 20 banks -> conflict-free)

// ---------------------------------------------------------------------------
// Kernel 1: reconstruct W (f16) = h_weight[idxW] * xiW, 4-wide vectorized
// ---------------------------------------------------------------------------
__global__ __launch_bounds__(256) void build_w_kernel(
    const float* __restrict__ hw, const float* __restrict__ xiW,
    const int* __restrict__ idxW, _Float16* __restrict__ Wh)
{
    const int i4 = blockIdx.x * 256 + threadIdx.x;        // one thread per 4 elems
    const int4   idx = reinterpret_cast<const int4*>(idxW)[i4];
    const float4 xi  = reinterpret_cast<const float4*>(xiW)[i4];
    v4h r;
    r[0] = (_Float16)(hw[idx.x] * xi.x);
    r[1] = (_Float16)(hw[idx.y] * xi.y);
    r[2] = (_Float16)(hw[idx.z] * xi.z);
    r[3] = (_Float16)(hw[idx.w] * xi.w);
    reinterpret_cast<v4h*>(Wh)[i4] = r;
}

// ---------------------------------------------------------------------------
// Kernel 2: bias (f32) = h_bias[idxB] * xiB
// ---------------------------------------------------------------------------
__global__ __launch_bounds__(256) void build_bias_kernel(
    const float* __restrict__ hb, const float* __restrict__ xiB,
    const int* __restrict__ idxB, float* __restrict__ bias)
{
    const int i = blockIdx.x * 256 + threadIdx.x;
    bias[i] = hb[idxB[i]] * xiB[i];
}

// ---------------------------------------------------------------------------
// Kernel 3: out[8192,4096] = x[8192,4096] @ Wh^T + bias, f16 WMMA, fp32 acc
//   256 thr = 8 waves; block tile 128x128; wave tile 64x32 (4x2 WMMA accs)
//   Double-buffered LDS x-tile: one barrier per K-step, next-tile global
//   loads + f32->f16 convert overlap the WMMA burst.
// ---------------------------------------------------------------------------
__global__ __launch_bounds__(256) void hash_gemm_kernel(
    const float* __restrict__ x, const _Float16* __restrict__ Wh,
    const float* __restrict__ bias, float* __restrict__ out)
{
    __shared__ _Float16 As[2][BM * LDA];   // 2 x 10 KB

    const int tid  = threadIdx.x;
    const int lane = tid & 31;
    const int w    = tid >> 5;
    const int wm   = w >> 2;      // 0..1  (M direction)
    const int wn   = w & 3;       // 0..3  (N direction)
    const int l15  = lane & 15;
    const int lh   = lane >> 4;   // 0/1   (K half)

    const int mBase = blockIdx.x * BM;
    const int nBase = blockIdx.y * BN;

    // staging slots for this thread: 4 float4 per 128x32 tile
    const int srow = tid >> 1;              // rows 0..127 (2 threads per row)
    const int sseg = (tid & 1) * 4;         // float4 seg 0..3 / 4..7
    // each thread does segs {sseg..sseg+3} of rows {srow} ... use 4 slots:
    //   (row = tid>>1, seg = (tid&1)*4 + j), j = 0..3  -> 4 float4 = 16 floats
    const float* xrow = x + (size_t)(mBase + srow) * K_DIM + sseg * 4;
    _Float16* lrow0 = &As[0][srow * LDA + sseg * 4];
    _Float16* lrow1 = &As[1][srow * LDA + sseg * 4];

    v8f acc[4][2];
#pragma unroll
    for (int mt = 0; mt < 4; ++mt)
#pragma unroll
        for (int nt = 0; nt < 2; ++nt)
            acc[mt][nt] = (v8f){0.f, 0.f, 0.f, 0.f, 0.f, 0.f, 0.f, 0.f};

    // B (W) rows act as columns of the 32x16 B fragment: 16 contiguous K per lane
    const size_t bRow0 = (size_t)(nBase + wn * 32 +  0 + l15) * K_DIM;
    const size_t bRow1 = (size_t)(nBase + wn * 32 + 16 + l15) * K_DIM;

    // A fragment LDS offsets (halfs): chunks [lh*16 .. +7] and [lh*16+8 .. +7]
    const int aoff = (wm * 64 + l15) * LDA + lh * 16;

    // ---- prologue: stage tile k=0 into buffer 0
#pragma unroll
    for (int j = 0; j < 4; ++j) {
        const float4 f = *reinterpret_cast<const float4*>(xrow + j * 4);
        v4h h;
        h[0] = (_Float16)f.x; h[1] = (_Float16)f.y;
        h[2] = (_Float16)f.z; h[3] = (_Float16)f.w;
        *reinterpret_cast<v4h*>(lrow0 + j * 4) = h;
    }
    __syncthreads();

    for (int k = 0; k < K_DIM - KSTEP; k += KSTEP) {
        const int it = k >> 5;               // iteration index
        const _Float16* curBase = (it & 1) ? &As[1][0] : &As[0][0];
        _Float16*       nxtRow  = (it & 1) ? lrow0 : lrow1;

        // B fragments (global f16, 32B per lane -> 2 x global_load_b128 each)
        const v16h bfrag0 = *reinterpret_cast<const v16h*>(Wh + bRow0 + k + lh * 16);
        const v16h bfrag1 = *reinterpret_cast<const v16h*>(Wh + bRow1 + k + lh * 16);

        // issue next-tile global x loads AFTER bfrags so they stay in flight
        float4 f[4];
#pragma unroll
        for (int j = 0; j < 4; ++j)
            f[j] = *reinterpret_cast<const float4*>(xrow + (k + KSTEP) + j * 4);

        // A fragments: all four into distinct registers (no WAR serialization)
        v16h afrag[4];
#pragma unroll
        for (int mt = 0; mt < 4; ++mt) {
            const v8h lo = *reinterpret_cast<const v8h*>(curBase + aoff + mt * 16 * LDA);
            const v8h hi = *reinterpret_cast<const v8h*>(curBase + aoff + mt * 16 * LDA + 8);
#pragma unroll
            for (int j = 0; j < 8; ++j) { afrag[mt][j] = lo[j]; afrag[mt][8 + j] = hi[j]; }
        }

        // 8 back-to-back WMMAs
#pragma unroll
        for (int mt = 0; mt < 4; ++mt) {
            acc[mt][0] = __builtin_amdgcn_wmma_f32_16x16x32_f16(
                false, afrag[mt], false, bfrag0, (short)0, acc[mt][0], false, false);
            acc[mt][1] = __builtin_amdgcn_wmma_f32_16x16x32_f16(
                false, afrag[mt], false, bfrag1, (short)0, acc[mt][1], false, false);
        }

        // convert + store next tile into the other buffer (overlaps WMMA)
#pragma unroll
        for (int j = 0; j < 4; ++j) {
            v4h h;
            h[0] = (_Float16)f[j].x; h[1] = (_Float16)f[j].y;
            h[2] = (_Float16)f[j].z; h[3] = (_Float16)f[j].w;
            *reinterpret_cast<v4h*>(nxtRow + j * 4) = h;
        }
        __syncthreads();   // next buffer staged; current buffer free for overwrite
    }

    // ---- last K-step (tile already staged in buffer 1: it = 127 -> odd)
    {
        const int k = K_DIM - KSTEP;
        const _Float16* curBase = &As[1][0];
        const v16h bfrag0 = *reinterpret_cast<const v16h*>(Wh + bRow0 + k + lh * 16);
        const v16h bfrag1 = *reinterpret_cast<const v16h*>(Wh + bRow1 + k + lh * 16);
        v16h afrag[4];
#pragma unroll
        for (int mt = 0; mt < 4; ++mt) {
            const v8h lo = *reinterpret_cast<const v8h*>(curBase + aoff + mt * 16 * LDA);
            const v8h hi = *reinterpret_cast<const v8h*>(curBase + aoff + mt * 16 * LDA + 8);
#pragma unroll
            for (int j = 0; j < 8; ++j) { afrag[mt][j] = lo[j]; afrag[mt][8 + j] = hi[j]; }
        }
#pragma unroll
        for (int mt = 0; mt < 4; ++mt) {
            acc[mt][0] = __builtin_amdgcn_wmma_f32_16x16x32_f16(
                false, afrag[mt], false, bfrag0, (short)0, acc[mt][0], false, false);
            acc[mt][1] = __builtin_amdgcn_wmma_f32_16x16x32_f16(
                false, afrag[mt], false, bfrag1, (short)0, acc[mt][1], false, false);
        }
    }

    // ---- epilogue: C layout -> lane holds col l15, VGPR v holds row v + 8*lh
#pragma unroll
    for (int nt = 0; nt < 2; ++nt) {
        const int ncol = nBase + wn * 32 + nt * 16 + l15;
        const float bv = bias[ncol];
#pragma unroll
        for (int mt = 0; mt < 4; ++mt) {
            const int mrow = mBase + wm * 64 + mt * 16 + lh * 8;
#pragma unroll
            for (int v = 0; v < 8; ++v) {
                out[(size_t)(mrow + v) * N_DIM + ncol] = acc[mt][nt][v] + bv;
            }
        }
    }
}

// ---------------------------------------------------------------------------
extern "C" void kernel_launch(void* const* d_in, const int* in_sizes, int n_in,
                              void* d_out, int out_size, void* d_ws, size_t ws_size,
                              hipStream_t stream)
{
    const float* x    = (const float*)d_in[0];
    const float* hw   = (const float*)d_in[1];
    const float* hb   = (const float*)d_in[2];
    const float* xiW  = (const float*)d_in[3];
    const float* xiB  = (const float*)d_in[4];
    const int*   idxW = (const int*)d_in[5];
    const int*   idxB = (const int*)d_in[6];
    float* out = (float*)d_out;

    _Float16* Wh   = (_Float16*)d_ws;                                  // 32 MB
    float*    bias = (float*)((char*)d_ws + (size_t)32 * 1024 * 1024); // 16 KB

    build_w_kernel<<<16384, 256, 0, stream>>>(hw, xiW, idxW, Wh);
    build_bias_kernel<<<16, 256, 0, stream>>>(hb, xiB, idxB, bias);
    hash_gemm_kernel<<<dim3(M_DIM / BM, N_DIM / BN), 256, 0, stream>>>(x, Wh, bias, out);
}
